// GConv_58961311039784
// MI455X (gfx1250) — matrix operand order repaired and checked by reading, not compiled, not run on hardware
//
#include <hip/hip_runtime.h>

typedef __bf16       v16bf __attribute__((ext_vector_type(16)));
typedef float        v8f   __attribute__((ext_vector_type(8)));
typedef unsigned int v8u   __attribute__((ext_vector_type(8)));
typedef unsigned int v4u   __attribute__((ext_vector_type(4)));
typedef int          v8i   __attribute__((ext_vector_type(8)));
typedef int          v4i   __attribute__((ext_vector_type(4)));

#if __has_builtin(__builtin_amdgcn_tensor_load_to_lds) && __has_builtin(__builtin_amdgcn_s_wait_tensorcnt)
#define USE_TDM 1
#else
#define USE_TDM 0
#endif

#define CIN    64
#define COUT   64
#define ROT    4
#define NTAP   9
#define HH     64
#define WW     64
#define NB     8
#define KTOT   256            // ROT*CIN
#define NKB    8              // K-blocks of 32
// x element strides (floats):  [n][c][r][h][w]
#define XS_N   (CIN*ROT*HH*WW)   // 1048576
#define XS_C   (ROT*HH*WW)       // 16384
#define XS_R   (HH*WW)           // 4096
// out element strides (floats): [n][576][4][64][64]
#define OS_CH  (ROT*HH*WW)       // 16384
// Wp: bf16, per (rot,tap) slice = [hi|lo], each part [kb(8)][ob(4)][lane(32)][i(16)]
#define WPART  16384             // bf16 elems per part per slice (32KB)
#define WSLICE (2*WPART)         // 64KB per slice
#define WP_BYTES (ROT*NTAP*WSLICE*2)          // 2,359,296
// xp: padded packed x, u32 {bf16_lo<<16 | bf16_hi}, [n][k(256)][66][66]
#define HP     66
#define XP_PLANE (HP*HP)                      // 4356
#define XP_ELEMS ((size_t)NB*KTOT*XP_PLANE)   // 8,921,088 u32
#define XP_BYTES (XP_ELEMS*4)                 // 35,684,352
#define WS_NEED (WP_BYTES + XP_BYTES)

static __device__ __forceinline__ unsigned short bf16_hi_bits(float v, float* rem) {
  __bf16 h = (__bf16)v;
  *rem = v - (float)h;
  return __builtin_bit_cast(unsigned short, h);
}

// ---------------------------------------------------------------------------
// Prep 1: rotation-permute weights, split fp32 -> bf16 (hi, lo), scatter into
// a V_WMMA_F32_16X16X32_BF16 A layout with contiguous K per lane half (same order as B):
//   lanes 0-15 : M=lane,    elem i -> K = i        (K 0..15)
//   lanes 16-31: M=lane-16, elem i -> K = i + 16   (K 16..31)
// with k = r*64 + c, rotation permutation r' = (r - rot) & 3.
// ---------------------------------------------------------------------------
__global__ __launch_bounds__(256) void prep_weights(const float* __restrict__ w,
                                                    __bf16* __restrict__ wp) {
  int idx = blockIdx.x * 256 + threadIdx.x;          // 36 * 16384 = 589824 total
  if (idx >= ROT * NTAP * WPART) return;
  int i    = idx & 15;
  int lane = (idx >> 4) & 31;
  int ob   = (idx >> 9) & 3;
  int kb   = (idx >> 11) & 7;
  int rt   = idx >> 14;                               // rot*9 + tap in [0,36)
  int tap  = rt % NTAP;
  int rot  = rt / NTAP;

  int k = kb * 32 + i + ((lane >> 4) << 4);
  int m = lane & 15;
  int o = ob * 16 + m;
  int r = k >> 6;
  int c = k & 63;
  int rsrc = (r - rot) & 3;
  // weight strides: o:64*4*9=2304  c:4*9=36  r:9  t:1
  float v = w[o * 2304 + c * 36 + rsrc * 9 + tap];
  float rem;
  unsigned short hb = bf16_hi_bits(v, &rem);
  __bf16 l = (__bf16)rem;
  int inner = idx & (WPART - 1);
  wp[(size_t)rt * WSLICE + inner]         = __builtin_bit_cast(__bf16, hb);
  wp[(size_t)rt * WSLICE + WPART + inner] = l;
}

// ---------------------------------------------------------------------------
// Prep 2: zero-padded, K-major, bf16-hi/lo-packed copy of x:
//   xp[n][k=r*64+c][hp(66)][wp(66)] : u32 = (bits(lo) << 16) | bits(hi)
// Border (hp==0|65, wp==0|65) = 0, so all tap-shifted reads are in-bounds.
// ---------------------------------------------------------------------------
__global__ __launch_bounds__(256) void prep_xpack(const float* __restrict__ x,
                                                  unsigned int* __restrict__ xp) {
  int plane = blockIdx.x / 18;                // n*256 + k
  int tid   = (blockIdx.x % 18) * 256 + threadIdx.x;
  if (tid >= XP_PLANE) return;
  int hp = tid / HP;
  int wpx = tid - hp * HP;
  int n = plane >> 8;
  int k = plane & 255;
  int r = k >> 6;
  int c = k & 63;
  int h = hp - 1, w_ = wpx - 1;
  float v = 0.0f;
  if (h >= 0 && h < HH && w_ >= 0 && w_ < WW)
    v = x[(size_t)n * XS_N + (size_t)c * XS_C + r * XS_R + h * WW + w_];
  float rem;
  unsigned short hb = bf16_hi_bits(v, &rem);
  unsigned short lb = __builtin_bit_cast(unsigned short, (__bf16)rem);
  xp[(size_t)plane * XP_PLANE + tid] = ((unsigned int)lb << 16) | hb;
}

// ---------------------------------------------------------------------------
// Stage one 64KB weight slice into LDS. Fast path: per-wave TDM descriptor
// (8 waves x 8KB 1-D tiles, 8-byte elements) + s_wait_tensorcnt.
// ---------------------------------------------------------------------------
static __device__ __forceinline__ void stage_weights(const __bf16* __restrict__ wsrc,
                                                     __bf16* lds_w, int wave) {
#if USE_TDM
  unsigned lds_base = (unsigned)(size_t)(void*)lds_w + (unsigned)wave * 8192u;
  unsigned long long ga = (unsigned long long)(size_t)(const void*)wsrc
                        + (unsigned long long)wave * 8192ull;
  v4u g0;
  g0[0] = 1u;                                          // count=1, user mode
  g0[1] = lds_base;                                    // lds_addr
  g0[2] = (unsigned)(ga & 0xffffffffu);                // global_addr[31:0]
  g0[3] = (unsigned)((ga >> 32) & 0x01ffffffu)         // global_addr[56:32]
        | (2u << 30);                                  // type=2 ("image")
  v8i g1;
  g1[0] = (int)(3u << 16);                             // data_size=3 (8B), mask=0
  g1[1] = (int)(1024u << 16);                          // tensor_dim0 = 1024 (low 16b at bit 48)
  g1[2] = (int)(1u << 16);                             // tensor_dim0 hi=0, tensor_dim1 = 1
  g1[3] = (int)(1024u << 16);                          // tensor_dim1 hi=0, tile_dim0 = 1024
  g1[4] = 1;                                           // tile_dim1 = 1, tile_dim2 = 0
  g1[5] = 1024;                                        // tensor_dim0_stride[31:0] = 1024
  g1[6] = 0;                                           // stride hi, dim1_stride lo
  g1[7] = 0;
  v4i z4 = {0, 0, 0, 0};
  v8i z8 = {0, 0, 0, 0, 0, 0, 0, 0};
  __builtin_amdgcn_tensor_load_to_lds(g0, g1, z4, z4, z8, 0);
  __builtin_amdgcn_s_wait_tensorcnt(0);
#else
  const uint4* src = (const uint4*)wsrc;
  uint4* dst = (uint4*)lds_w;
#pragma unroll
  for (int i = 0; i < 16; ++i)
    dst[threadIdx.x + i * 256] = src[threadIdx.x + i * 256];
  (void)wave;
#endif
}

// ---------------------------------------------------------------------------
// Main kernel (fast path): one WG (256 thr = 8 waves) per (n, rot, tap, 2-row
// h-tile). Per K-block of 32: 16 packed loads/lane, 16 v_perm_b32 to build
// Bh/Bl, A hi/lo from LDS, D += Ah*Bh + Ah*Bl + Al*Bh  (bf16x3 fp32 emulation).
// B layout: lanes 0-15 K=0..15, lanes 16-31 K=16..31, elems sequential.
// ---------------------------------------------------------------------------
__global__ __launch_bounds__(256) void gconv_wmma_pk(const unsigned int* __restrict__ xp,
                                                     const __bf16* __restrict__ wp,
                                                     float* __restrict__ out) {
  __shared__ __bf16 lds_w[WSLICE];                   // 64 KB: [hi 32KB | lo 32KB]

  int b = blockIdx.x;
  int htile = b & 31;  b >>= 5;
  int tap   = b % NTAP; b /= NTAP;
  int rot   = b & 3;
  int n     = b >> 2;
  int rt    = rot * NTAP + tap;

  int wave = threadIdx.x >> 5;
  int lane = threadIdx.x & 31;

  stage_weights(wp + (size_t)rt * WSLICE, lds_w, wave);
  __syncthreads();

  int h  = htile * 2 + (wave >> 2);
  int w0 = (wave & 3) * 16;

  const int ring_dy[8] = {-1, -1, -1, 0, 1, 1, 1, 0};
  const int ring_dx[8] = {-1, 0, 1, 1, 1, 0, -1, -1};
  int dy = 0, dx = 0;
  if (tap != 0) {
    int ri = (tap - 1 + 2 * rot) & 7;
    dy = ring_dy[ri];
    dx = ring_dx[ri];
  }

  // padded coordinates: always in-bounds, no masking needed
  int hyp = h + dy + 1;                       // [0,65]
  int wxp = w0 + (lane & 15) + dx + 1;        // [0,65]
  int c0  = (lane >> 4) * 16;                 // lane half -> K offset 0/16
  const unsigned int* xb = xp + (size_t)(n * KTOT + c0) * XP_PLANE + hyp * HP + wxp;

  const v16bf* lw = (const v16bf*)lds_w;      // hi: [(kb*4+ob)*32+lane]; lo: +1024

  v8f acc0 = {}, acc1 = {}, acc2 = {}, acc3 = {};

#pragma unroll
  for (int kb = 0; kb < NKB; ++kb) {
    // ---- 16 packed {lo,hi} words per lane (consecutive K) ----
    unsigned int xw[16];
#pragma unroll
    for (int i = 0; i < 16; ++i)
      xw[i] = xb[(kb * 32 + i) * XP_PLANE];

    // ---- split halves: VGPR j of B = {elem 2j, elem 2j+1} ----
    v8u bhw, blw;
#pragma unroll
    for (int j = 0; j < 8; ++j) {
      bhw[j] = __builtin_amdgcn_perm(xw[2 * j + 1], xw[2 * j], 0x05040100u); // hi bf16s
      blw[j] = __builtin_amdgcn_perm(xw[2 * j + 1], xw[2 * j], 0x07060302u); // lo bf16s
    }
    v16bf bh = __builtin_bit_cast(v16bf, bhw);
    v16bf bl = __builtin_bit_cast(v16bf, blw);

    // ---- 4 o-blocks x 3 WMMA terms ----
#pragma unroll
    for (int ob = 0; ob < 4; ++ob) {
      v16bf ah = lw[(kb * 4 + ob) * 32 + lane];
      v16bf al = lw[1024 + (kb * 4 + ob) * 32 + lane];
      v8f* acc = (ob == 0) ? &acc0 : (ob == 1) ? &acc1 : (ob == 2) ? &acc2 : &acc3;
      *acc = __builtin_amdgcn_wmma_f32_16x16x32_bf16(false, ah, false, bh, (short)0, *acc, false, false);
      *acc = __builtin_amdgcn_wmma_f32_16x16x32_bf16(false, ah, false, bl, (short)0, *acc, false, false);
      *acc = __builtin_amdgcn_wmma_f32_16x16x32_bf16(false, al, false, bh, (short)0, *acc, false, false);
    }
  }

  int o_hi = (lane >> 4) * 8;
  float* op = out + (size_t)(((n * 576 + tap * 64 + o_hi) * ROT + rot)) * XS_R
                  + (size_t)h * WW + w0 + (lane & 15);
#pragma unroll
  for (int j = 0; j < 8; ++j) {
    op[(0 * 16 + j) * OS_CH] = acc0[j];
    op[(1 * 16 + j) * OS_CH] = acc1[j];
    op[(2 * 16 + j) * OS_CH] = acc2[j];
    op[(3 * 16 + j) * OS_CH] = acc3[j];
  }
}

// ---------------------------------------------------------------------------
// Fallback (small workspace): convert fp32 x -> bf16 hi/lo in-register.
// ---------------------------------------------------------------------------
__global__ __launch_bounds__(256) void gconv_wmma_cv(const float* __restrict__ x,
                                                     const __bf16* __restrict__ wp,
                                                     float* __restrict__ out) {
  __shared__ __bf16 lds_w[WSLICE];

  int b = blockIdx.x;
  int htile = b & 31;  b >>= 5;
  int tap   = b % NTAP; b /= NTAP;
  int rot   = b & 3;
  int n     = b >> 2;
  int rt    = rot * NTAP + tap;

  {
    const uint4* src = (const uint4*)(wp + (size_t)rt * WSLICE);
    uint4* dst = (uint4*)lds_w;
#pragma unroll
    for (int i = 0; i < 16; ++i)
      dst[threadIdx.x + i * 256] = src[threadIdx.x + i * 256];
  }
  __syncthreads();

  int wave = threadIdx.x >> 5;
  int lane = threadIdx.x & 31;
  int h  = htile * 2 + (wave >> 2);
  int w0 = (wave & 3) * 16;

  const int ring_dy[8] = {-1, -1, -1, 0, 1, 1, 1, 0};
  const int ring_dx[8] = {-1, 0, 1, 1, 1, 0, -1, -1};
  int dy = 0, dx = 0;
  if (tap != 0) {
    int ri = (tap - 1 + 2 * rot) & 7;
    dy = ring_dy[ri];
    dx = ring_dx[ri];
  }

  int hy = h + dy;
  int wx = w0 + (lane & 15) + dx;
  bool valid = (hy >= 0) && (hy < HH) && (wx >= 0) && (wx < WW);
  int hyc = hy < 0 ? 0 : (hy > HH - 1 ? HH - 1 : hy);
  int wxc = wx < 0 ? 0 : (wx > WW - 1 ? WW - 1 : wx);

  int c0 = (lane >> 4) * 16;
  const float* xb = x + (size_t)n * XS_N + (size_t)c0 * XS_C + hyc * WW + wxc;
  const v16bf* lw = (const v16bf*)lds_w;

  v8f acc0 = {}, acc1 = {}, acc2 = {}, acc3 = {};

#pragma unroll
  for (int kb = 0; kb < NKB; ++kb) {
    float xv[16];
#pragma unroll
    for (int i = 0; i < 16; ++i)
      xv[i] = xb[((kb & 1) * 32 + i) * XS_C + (kb >> 1) * XS_R];

    v16bf bh, bl;
#pragma unroll
    for (int i = 0; i < 16; ++i) {
      float v = valid ? xv[i] : 0.0f;
      __bf16 hv = (__bf16)v;
      bh[i] = hv;
      bl[i] = (__bf16)(v - (float)hv);
    }

#pragma unroll
    for (int ob = 0; ob < 4; ++ob) {
      v16bf ah = lw[(kb * 4 + ob) * 32 + lane];
      v16bf al = lw[1024 + (kb * 4 + ob) * 32 + lane];
      v8f* acc = (ob == 0) ? &acc0 : (ob == 1) ? &acc1 : (ob == 2) ? &acc2 : &acc3;
      *acc = __builtin_amdgcn_wmma_f32_16x16x32_bf16(false, ah, false, bh, (short)0, *acc, false, false);
      *acc = __builtin_amdgcn_wmma_f32_16x16x32_bf16(false, ah, false, bl, (short)0, *acc, false, false);
      *acc = __builtin_amdgcn_wmma_f32_16x16x32_bf16(false, al, false, bh, (short)0, *acc, false, false);
    }
  }

  int o_hi = (lane >> 4) * 8;
  float* op = out + (size_t)(((n * 576 + tap * 64 + o_hi) * ROT + rot)) * XS_R
                  + (size_t)h * WW + w0 + (lane & 15);
#pragma unroll
  for (int j = 0; j < 8; ++j) {
    op[(0 * 16 + j) * OS_CH] = acc0[j];
    op[(1 * 16 + j) * OS_CH] = acc1[j];
    op[(2 * 16 + j) * OS_CH] = acc2[j];
    op[(3 * 16 + j) * OS_CH] = acc3[j];
  }
}

// ---------------------------------------------------------------------------
extern "C" void kernel_launch(void* const* d_in, const int* in_sizes, int n_in,
                              void* d_out, int out_size, void* d_ws, size_t ws_size,
                              hipStream_t stream) {
  const float* x = (const float*)d_in[0];   // [8,64,4,64,64] fp32
  const float* w = (const float*)d_in[1];   // [64,64,4,9] fp32
  float* out = (float*)d_out;               // [8,576,4,64,64] fp32
  __bf16* wpks = (__bf16*)d_ws;

  (void)in_sizes; (void)n_in; (void)out_size;

  prep_weights<<<(ROT * NTAP * WPART + 255) / 256, 256, 0, stream>>>(w, wpks);

  if (ws_size >= (size_t)WS_NEED) {
    unsigned int* xpk = (unsigned int*)((char*)d_ws + WP_BYTES);
    prep_xpack<<<NB * KTOT * 18, 256, 0, stream>>>(x, xpk);
    gconv_wmma_pk<<<NB * ROT * NTAP * 32, 256, 0, stream>>>(xpk, wpks, out);
  } else {
    gconv_wmma_cv<<<NB * ROT * NTAP * 32, 256, 0, stream>>>(x, wpks, out);
  }
}
